// AFTAttention_77592879169591
// MI455X (gfx1250) — matrix operand order repaired and checked by reading, not compiled, not run on hardware
//
#include <hip/hip_runtime.h>
#include <hip/hip_bf16.h>

typedef _Float16 v8h  __attribute__((ext_vector_type(8)));
typedef _Float16 v16h __attribute__((ext_vector_type(16)));
typedef float    v8f  __attribute__((ext_vector_type(8)));

#define NB   16          // batch
#define NC   192         // channels
#define HH   64
#define WW   64
#define HWSZ 4096
#define KS   31
#define PADW 112         // 64 + 48 (covers j0+63 window), rows padded to 94
#define PADH 94

// LDS layout (halves): kpad[94*112] | kvpad[94*112] | Bbuf[31*1024] | floats...
#define KPAD_OFF  0
#define KVPAD_OFF (PADH*PADW)              // 10528
#define BBUF_OFF  (2*PADH*PADW)            // 21056
#define BBUF_HALVES (31*1024)              // per u: blk0[16n x 32k], blk1[16n x 32k]
#define FLOAT_BYTE_OFF ((BBUF_OFF + BBUF_HALVES) * 2)   // 105600
#define LDS_BYTES (FLOAT_BYTE_OFF + 961*4 + 256*4 + 256*4 + 16)  // ~111.5 KB

__device__ __forceinline__ v16h load_frag(const _Float16* p) {
    v8h lo = *(const v8h*)p;
    v8h hi = *(const v8h*)(p + 16);
    return __builtin_shufflevector(lo, hi, 0,1,2,3,4,5,6,7,8,9,10,11,12,13,14,15);
}

#define WMMA_ACC(a, b, c) __builtin_amdgcn_wmma_f32_16x16x32_f16(false,(a),false,(b),(short)0,(c),false,false)

__global__ __launch_bounds__(256, 1)
void aft_attn_kernel(const float* __restrict__ qkv,
                     const float* __restrict__ wconv,
                     float* __restrict__ out) {
    extern __shared__ float4 smem4[];
    char* smem = (char*)smem4;
    _Float16* kpad  = (_Float16*)smem + KPAD_OFF;
    _Float16* kvpad = (_Float16*)smem + KVPAD_OFF;
    _Float16* Bbuf  = (_Float16*)smem + BBUF_OFF;
    float* wexp = (float*)(smem + FLOAT_BYTE_OFF);   // 961
    float* red  = wexp + 961;                        // 256
    float* red2 = red + 256;                         // 256

    const int tid = threadIdx.x;
    const int bc  = blockIdx.x;
    const int b   = bc / NC;
    const int c   = bc - b * NC;

    const float* qg = qkv + ((size_t)b * (3*NC) + c) * HWSZ;
    const float* kg = qg + (size_t)NC * HWSZ;
    const float* vg = qg + (size_t)(2*NC) * HWSZ;

    // ---- zero both padded maps (as dwords) ----
    {
        uint32_t* z = (uint32_t*)smem;
        for (int i = tid; i < PADH*PADW; i += 256) z[i] = 0u;   // 2 maps * 10528 halves / 2
    }
    // ---- expm1(weights) for this channel ----
    for (int i = tid; i < KS*KS; i += 256) wexp[i] = expm1f(wconv[(size_t)c * (KS*KS) + i]);
    __syncthreads();

    // ---- build banded Toeplitz B blocks: Bbuf[u][blk][n][k] = w[u, blk*32 + k - n] ----
    for (int e = tid; e < BBUF_HALVES; e += 256) {
        int u   = e >> 10;
        int rem = e & 1023;
        int blk = rem >> 9;
        int n   = (rem >> 5) & 15;
        int s   = rem & 31;
        int t   = blk * 32 + s - n;
        float v = (t >= 0 && t <= 30) ? wexp[u * KS + t] : 0.0f;
        Bbuf[e] = (_Float16)v;
    }

    // ---- softmax (unnormalized exp) over the 4096-element map ----
    float kreg[16];
    float lmax = -3.4e38f;
    #pragma unroll
    for (int i = 0; i < 16; ++i) {
        int p = tid + (i << 8);
        float x = kg[p];
        kreg[i] = x;
        lmax = fmaxf(lmax, x);
    }
    red[tid] = lmax;
    __syncthreads();
    for (int off = 128; off > 0; off >>= 1) {
        if (tid < off) red[tid] = fmaxf(red[tid], red[tid + off]);
        __syncthreads();
    }
    const float mk = red[0];
    __syncthreads();

    float s1 = 0.f, s2 = 0.f;
    #pragma unroll
    for (int i = 0; i < 16; ++i) {
        int p  = tid + (i << 8);
        float ek  = __expf(kreg[i] - mk);
        float ekv = ek * vg[p];
        s1 += ek; s2 += ekv;
        int row = p >> 6, col = p & 63;
        int idx = (row + 15) * PADW + (col + 15);
        kpad[idx]  = (_Float16)ek;
        kvpad[idx] = (_Float16)ekv;
    }
    red[tid] = s1; red2[tid] = s2;
    __syncthreads();
    for (int off = 128; off > 0; off >>= 1) {
        if (tid < off) { red[tid] += red[tid + off]; red2[tid] += red2[tid + off]; }
        __syncthreads();
    }
    const float sumK  = red[0];
    const float sumKV = red2[0];

    // ---- WMMA conv: y = sum_u ShiftRows_u(X) @ Toeplitz_u ----
    const int wave = tid >> 5;
    const int lane = tid & 31;
    const int jn   = wave & 3;         // N tile (output cols j0..j0+15)
    const int mh   = wave >> 2;        // row half: rows mh*32 .. mh*32+31
    const int j0   = jn << 4;
    const int n    = lane & 15;        // A: M row / B: N col / D: N col
    const int hi   = lane >> 4;        // half-wave select
    const int koffA = hi << 3;         // A K offset 0 / 8

    v8f accK0 = {}, accK1 = {}, accKV0 = {}, accKV1 = {};

    for (int u = 0; u < KS; ++u) {
        const _Float16* Bu = Bbuf + (u << 10);
        v16h b0 = load_frag(Bu +       (n << 5) + (hi << 3));   // blk0 band
        v16h b1 = load_frag(Bu + 512 + (n << 5) + (hi << 3));   // blk1 band

        const int rb = mh * 32 + n + u;          // padded row for M-tile 0
        const _Float16* k0r  = kpad  + rb * PADW + j0 + koffA;
        const _Float16* k1r  = k0r  + 16 * PADW;
        const _Float16* kv0r = kvpad + rb * PADW + j0 + koffA;
        const _Float16* kv1r = kv0r + 16 * PADW;

        v16h a;
        a = load_frag(k0r);        accK0  = WMMA_ACC(a, b0, accK0);
        a = load_frag(k0r + 32);   accK0  = WMMA_ACC(a, b1, accK0);
        a = load_frag(k1r);        accK1  = WMMA_ACC(a, b0, accK1);
        a = load_frag(k1r + 32);   accK1  = WMMA_ACC(a, b1, accK1);
        a = load_frag(kv0r);       accKV0 = WMMA_ACC(a, b0, accKV0);
        a = load_frag(kv0r + 32);  accKV0 = WMMA_ACC(a, b1, accKV0);
        a = load_frag(kv1r);       accKV1 = WMMA_ACC(a, b0, accKV1);
        a = load_frag(kv1r + 32);  accKV1 = WMMA_ACC(a, b1, accKV1);
    }

    // ---- combine: out = sigmoid(q) * (conv(ekv)+sumKV) / (conv(ek)+sumK) ----
    float* og = out + ((size_t)b * NC + c) * HWSZ;
    #pragma unroll
    for (int r = 0; r < 8; ++r) {
        // D layout: lane<16 -> N=lane, M=r ; lane>=16 -> N=lane-16, M=r+8
        int i0 = mh * 32 + (hi << 3) + r;       // M-tile 0 rows
        int i1 = i0 + 16;                       // M-tile 1 rows
        {
            int p = (i0 << 6) + j0 + n;
            float sig = 1.0f / (1.0f + __expf(-qg[p]));
            og[p] = sig * (accKV0[r] + sumKV) / (accK0[r] + sumK);
        }
        {
            int p = (i1 << 6) + j0 + n;
            float sig = 1.0f / (1.0f + __expf(-qg[p]));
            og[p] = sig * (accKV1[r] + sumKV) / (accK1[r] + sumK);
        }
    }
}

extern "C" void kernel_launch(void* const* d_in, const int* in_sizes, int n_in,
                              void* d_out, int out_size, void* d_ws, size_t ws_size,
                              hipStream_t stream) {
    const float* qkv   = (const float*)d_in[0];
    const float* wconv = (const float*)d_in[1];
    float* out = (float*)d_out;
    (void)in_sizes; (void)n_in; (void)d_ws; (void)ws_size; (void)out_size;

    // dynamic LDS > 64KB: opt in every call (cheap, capture-safe host-side state)
    hipFuncSetAttribute((const void*)aft_attn_kernel,
                        hipFuncAttributeMaxDynamicSharedMemorySize, LDS_BYTES);

    aft_attn_kernel<<<NB * NC, 256, LDS_BYTES, stream>>>(qkv, wconv, out);
}
